// CRSDBlock_71339406786972
// MI455X (gfx1250) — matrix operand order repaired and checked by reading, not compiled, not run on hardware
//
#include <hip/hip_runtime.h>
#include <hip/hip_bf16.h>
#include <math.h>

// ---------------------------------------------------------------------------
// CRSD block for MI455X (gfx1250, wave32).
//   T=8192 timesteps, D=1024, L=2 layers, K=2 reservoirs, alpha=0.1
//
//   1) Xp[t] = Wx @ x_t + b   -> batched over t: bf16 WMMA GEMM
//                                (v_wmma_f32_16x16x32_bf16, 8 WMMA / K-step / wave)
//   2) serial recurrence      -> persistent 64-WG kernel, recurrent weights
//      pinned in LDS as bf16 (160KB/WGP); h/r staged each step with
//      GLOBAL_LOAD_ASYNC_TO_LDS_B128 (ASYNCcnt) when available;
//      grid-wide epoch barrier twice per step.
// ---------------------------------------------------------------------------

#define T_STEPS 8192
#define DIM     1024
#define NLAYER  2
#define KRES    2
#define ALPHA_F 0.1f

#define NWG   64      // persistent workgroups for the recurrence
#define ROWS  16      // DIM / NWG rows owned per workgroup
#define TPB   256     // 8 waves of 32

typedef __attribute__((ext_vector_type(16))) __bf16 v16bf;
typedef __attribute__((ext_vector_type(8)))  __bf16 v8bf;
typedef __attribute__((ext_vector_type(8)))  float  v8f;
typedef __attribute__((ext_vector_type(4)))  float  f32x4;
typedef __attribute__((ext_vector_type(4)))  int    v4i;

// address-space-qualified int4 for the async-DMA builtin's pointer params
typedef __attribute__((address_space(1))) v4i as1_v4i;   // global
typedef __attribute__((address_space(3))) v4i as3_v4i;   // LDS

// --------------------------- async global -> LDS ----------------------------
// CDNA5 path: one B128 async DMA per lane (tracked by ASYNCcnt), bypassing
// VGPRs. Fallback: plain 16B load/store through registers.
__device__ __forceinline__ void async_g2l_b128(const float* g, float* l) {
#if defined(__gfx1250__) && __has_builtin(__builtin_amdgcn_global_load_async_to_lds_b128)
  as1_v4i* gp = (as1_v4i*)(void*)(g);
  as3_v4i* lp = (as3_v4i*)(void*)(l);
  __builtin_amdgcn_global_load_async_to_lds_b128(gp, lp, 0, 0);
#else
  *(f32x4*)l = *(const f32x4*)g;
#endif
}

__device__ __forceinline__ void wait_async0() {
#if defined(__gfx1250__) && __has_builtin(__builtin_amdgcn_global_load_async_to_lds_b128)
#if __has_builtin(__builtin_amdgcn_s_wait_asynccnt)
  __builtin_amdgcn_s_wait_asynccnt(0);
#else
  asm volatile("s_wait_asynccnt 0x0" ::: "memory");
#endif
#endif
}

// --------------------------- fp32 -> bf16 convert ---------------------------
__global__ void __launch_bounds__(256)
f32_to_bf16_kernel(const float* __restrict__ in, __bf16* __restrict__ out, int n) {
  int i = blockIdx.x * 256 + threadIdx.x;
  if (i < n) out[i] = (__bf16)in[i];
}

// --------------------------- WMMA GEMM: C = X @ W^T + bias ------------------
// X: [T,D] bf16 row-major, W: [D,D] bf16 row-major (Wx), C: [T,D] f32.
// Workgroup tile 128x128 (8 waves as 4x2); each wave 32x64 = 2x4 WMMA tiles,
// so 6 fragment loads feed 8 v_wmma per K-step.
__global__ void __launch_bounds__(256)
gemm_xw_bias_wmma(const __bf16* __restrict__ X, const __bf16* __restrict__ W,
                  const float* __restrict__ bias, float* __restrict__ C) {
  const int lane = threadIdx.x & 31;
  const int wave = threadIdx.x >> 5;          // 0..7
  const int wm = wave >> 1;                   // 0..3  (M direction)
  const int wn = wave & 1;                    // 0..1  (N direction)
  const int m0 = blockIdx.x * 128 + wm * 32;
  const int n0 = blockIdx.y * 128 + wn * 64;

  const int mrow = lane & 15;                 // M (for A) / N (for B) within tile
  const int half = lane >> 4;                 // lane-half selects K sub-range

  v8f acc[2][4] = {};

  for (int k0 = 0; k0 < DIM; k0 += 32) {
    // A fragments (16x32 bf16): VGPR0-3 <- K = k0+half*16+{0..7},
    //                           VGPR4-7 <- K = k0+half*16+8+{0..7}
    v16bf afrag[2];
#pragma unroll
    for (int mi = 0; mi < 2; ++mi) {
      const __bf16* pa = X + (size_t)(m0 + mi * 16 + mrow) * DIM + k0;
      v8bf lo = *(const v8bf*)(pa + half * 16);
      v8bf hi = *(const v8bf*)(pa + half * 16 + 8);
      v16bf a;
#pragma unroll
      for (int i = 0; i < 8; ++i) { a[i] = lo[i]; a[8 + i] = hi[i]; }
      afrag[mi] = a;
    }
    // B fragments (32x16 bf16): B[k][n] = W[n][k]; each lane: 16 contiguous K.
    v16bf bfrag[4];
#pragma unroll
    for (int ni = 0; ni < 4; ++ni) {
      const __bf16* pb = W + (size_t)(n0 + ni * 16 + mrow) * DIM + k0 + half * 16;
      bfrag[ni] = *(const v16bf*)pb;
    }
#pragma unroll
    for (int mi = 0; mi < 2; ++mi)
#pragma unroll
      for (int ni = 0; ni < 4; ++ni)
        acc[mi][ni] = __builtin_amdgcn_wmma_f32_16x16x32_bf16(
            false, afrag[mi], false, bfrag[ni], (short)0, acc[mi][ni], false, false);
  }

  // C/D layout: VGPR v -> row m0+mi*16 + v + half*8, col n0+ni*16+(lane&15)
#pragma unroll
  for (int mi = 0; mi < 2; ++mi)
#pragma unroll
    for (int ni = 0; ni < 4; ++ni) {
      const int col = n0 + ni * 16 + (lane & 15);
      const float bv = bias[col];
#pragma unroll
      for (int v = 0; v < 8; ++v) {
        const int row = m0 + mi * 16 + v + half * 8;
        C[(size_t)row * DIM + col] = acc[mi][ni][v] + bv;
      }
    }
}

// --------------------------- grid-wide epoch barrier ------------------------
__device__ inline void gsync(unsigned* cnt, unsigned target) {
  __syncthreads();
  if (threadIdx.x == 0) {
    __threadfence();                      // publish global writes
    atomicAdd(cnt, 1u);
    while (atomicAdd(cnt, 0u) < target) __builtin_amdgcn_s_sleep(2);
    __threadfence();                      // acquire others' writes
  }
  __syncthreads();
}

// --------------------------- persistent recurrence --------------------------
// Each of NWG workgroups owns ROWS rows of Wh, Wr0, Wr1, U0, U1, staged once
// into LDS as bf16 (5*16*1024*2 = 160KB). Per step:
//   phase A: pre = Xp[t] + Wh h + Wr0 r0 + Wr1 r1 ; h' = tanh(pre)   (sync)
//   phase B: r_k' = (1-a) r_k + a tanh(U_k h')                        (sync)
__global__ void __launch_bounds__(TPB)
crsd_recurrent(const float* __restrict__ Xp,   // [T,D] = Wx x + b
               const float* __restrict__ Wh,   // [D,D]
               const float* __restrict__ Wr,   // [K,D,D]
               const float* __restrict__ U,    // [K,D,D]
               float* __restrict__ out,        // [T,D]
               float* __restrict__ hbuf,       // [2,D] double-buffered h
               float* __restrict__ rbuf,       // [K,D]
               unsigned* __restrict__ sync) {
  extern __shared__ char smem[];
  __bf16* wlds = (__bf16*)smem;                        // 5*ROWS*DIM bf16
  float*  hl   = (float*)(smem + 5 * ROWS * DIM * 2);  // DIM f32
  float*  rl   = hl + DIM;                             // KRES*DIM f32

  const int wgrow0 = blockIdx.x * ROWS;
  const int tid = threadIdx.x;

  // Stage this WG's weight rows (fp32 global -> bf16 LDS), reused all T steps.
  const float* srcs[5] = {
      Wh + (size_t)wgrow0 * DIM,
      Wr + (size_t)wgrow0 * DIM,
      Wr + (size_t)DIM * DIM + (size_t)wgrow0 * DIM,
      U  + (size_t)wgrow0 * DIM,
      U  + (size_t)DIM * DIM + (size_t)wgrow0 * DIM };
  for (int m = 0; m < 5; ++m) {
    const float* s = srcs[m];
    __bf16* d = wlds + (size_t)m * ROWS * DIM;
    for (int i = tid; i < ROWS * DIM; i += TPB) d[i] = (__bf16)s[i];
  }
  __syncthreads();

  const int row  = tid >> 4;   // 0..15: local row
  const int c    = tid & 15;   // 16 lanes cooperate per row
  const int grow = wgrow0 + row;

  const __bf16* pWh  = wlds + (size_t)0 * ROWS * DIM + (size_t)row * DIM;
  const __bf16* pWr0 = wlds + (size_t)1 * ROWS * DIM + (size_t)row * DIM;
  const __bf16* pWr1 = wlds + (size_t)2 * ROWS * DIM + (size_t)row * DIM;
  const __bf16* pU0  = wlds + (size_t)3 * ROWS * DIM + (size_t)row * DIM;
  const __bf16* pU1  = wlds + (size_t)4 * ROWS * DIM + (size_t)row * DIM;

  float r_old0 = 0.f, r_old1 = 0.f;   // this WG owns its r rows across steps
  unsigned epoch = 0;

  for (int t = 0; t < T_STEPS; ++t) {
    const int cur = t & 1, nxt = cur ^ 1;

    // stage h_t (4KB: one async B128/lane) and r_t (8KB: two) into LDS
    async_g2l_b128(hbuf + cur * DIM + tid * 4, hl + tid * 4);
    async_g2l_b128(rbuf + tid * 4,            rl + tid * 4);
    async_g2l_b128(rbuf + DIM + tid * 4,      rl + DIM + tid * 4);
    wait_async0();
    __syncthreads();

    // ---- phase A ----
    float acc = 0.f;
#pragma unroll 8
    for (int j = c; j < DIM; j += 16)
      acc += (float)pWh[j] * hl[j] + (float)pWr0[j] * rl[j] + (float)pWr1[j] * rl[DIM + j];
    acc += __shfl_xor(acc, 1); acc += __shfl_xor(acc, 2);
    acc += __shfl_xor(acc, 4); acc += __shfl_xor(acc, 8);
    if (c == 0) {
      float hnew = tanhf(Xp[(size_t)t * DIM + grow] + acc);
      hbuf[nxt * DIM + grow] = hnew;
      out[(size_t)t * DIM + grow] = hnew;
      if (t + 1 < T_STEPS)
        __builtin_prefetch(&Xp[(size_t)(t + 1) * DIM + grow], 0, 0);
    }
    gsync(sync, ++epoch * NWG);

    // stage h_{t+1}
    async_g2l_b128(hbuf + nxt * DIM + tid * 4, hl + tid * 4);
    wait_async0();
    __syncthreads();

    // ---- phase B ----
    float a0 = 0.f, a1 = 0.f;
#pragma unroll 8
    for (int j = c; j < DIM; j += 16) {
      float hv = hl[j];
      a0 += (float)pU0[j] * hv;
      a1 += (float)pU1[j] * hv;
    }
    a0 += __shfl_xor(a0, 1); a0 += __shfl_xor(a0, 2);
    a0 += __shfl_xor(a0, 4); a0 += __shfl_xor(a0, 8);
    a1 += __shfl_xor(a1, 1); a1 += __shfl_xor(a1, 2);
    a1 += __shfl_xor(a1, 4); a1 += __shfl_xor(a1, 8);
    if (c == 0) {
      r_old0 = (1.f - ALPHA_F) * r_old0 + ALPHA_F * tanhf(a0);
      r_old1 = (1.f - ALPHA_F) * r_old1 + ALPHA_F * tanhf(a1);
      rbuf[grow]       = r_old0;
      rbuf[DIM + grow] = r_old1;
    }
    gsync(sync, ++epoch * NWG);
  }
}

// ---------------------------------------------------------------------------
extern "C" void kernel_launch(void* const* d_in, const int* in_sizes, int n_in,
                              void* d_out, int out_size, void* d_ws, size_t ws_size,
                              hipStream_t stream) {
  const float* x_seq = (const float*)d_in[0];            // [T,D]
  const float* Wx    = (const float*)d_in[1];            // [L,D,D]
  const float* Wh    = (const float*)d_in[2];            // [L,D,D]
  const float* Wr    = (const float*)d_in[3];            // [L,K,D,D]
  const float* U     = (const float*)d_in[4];            // [L,K,D,D]
  const float* b     = (const float*)d_in[5];            // [L,D]

  char* ws = (char*)d_ws;
  const size_t TD = (size_t)T_STEPS * DIM;
  __bf16*   Xb   = (__bf16*)(ws + 0);                    // 16 MB
  __bf16*   Wb   = (__bf16*)(ws + 16777216);             //  2 MB
  float*    Xp   = (float*) (ws + 18874368);             // 32 MB
  float*    seq0 = (float*) (ws + 52428800);             // 32 MB (layer0 out)
  float*    hbuf = (float*) (ws + 85983232);             //  8 KB
  float*    rbuf = (float*) (ws + 85991424);             //  8 KB
  unsigned* sync = (unsigned*)(ws + 85999616);           //  counter

  const int smem_bytes = 5 * ROWS * DIM * 2 + DIM * 4 + KRES * DIM * 4; // 176128
  (void)hipFuncSetAttribute((const void*)crsd_recurrent,
                            hipFuncAttributeMaxDynamicSharedMemorySize, smem_bytes);

  const dim3 gemm_grid(T_STEPS / 128, DIM / 128);

  for (int l = 0; l < NLAYER; ++l) {
    const float* src = (l == 0) ? x_seq : seq0;
    float* outseq    = (l == 0) ? seq0 : (float*)d_out;

    // 1) down-convert activations + Wx[l] to bf16
    f32_to_bf16_kernel<<<(int)(TD / 256), 256, 0, stream>>>(src, Xb, (int)TD);
    f32_to_bf16_kernel<<<(DIM * DIM) / 256, 256, 0, stream>>>(
        Wx + (size_t)l * DIM * DIM, Wb, DIM * DIM);

    // 2) Xp = X @ Wx^T + b  (WMMA bf16, fp32 accumulate)
    gemm_xw_bias_wmma<<<gemm_grid, 256, 0, stream>>>(Xb, Wb, b + (size_t)l * DIM, Xp);

    // 3) persistent serial recurrence
    (void)hipMemsetAsync(hbuf, 0, 2 * DIM * sizeof(float), stream);
    (void)hipMemsetAsync(rbuf, 0, KRES * DIM * sizeof(float), stream);
    (void)hipMemsetAsync(sync, 0, 256, stream);
    crsd_recurrent<<<NWG, TPB, smem_bytes, stream>>>(
        Xp,
        Wh + (size_t)l * DIM * DIM,
        Wr + (size_t)l * KRES * DIM * DIM,
        U  + (size_t)l * KRES * DIM * DIM,
        outseq, hbuf, rbuf, sync);
  }
}
